// HGTEncoder_50199577756186
// MI455X (gfx1250) — matrix operand order repaired and checked by reading, not compiled, not run on hardware
//
#include <hip/hip_runtime.h>
#include <hip/hip_bf16.h>
#include <math.h>

// ---------------------------------------------------------------------------
// HGT encoder for MI455X (gfx1250, wave32, WMMA).
// - Dense GEMMs: v_wmma_f32_16x16x32_f16, f32 accum. Weights pre-packed once
//   into f16 WMMA-B fragment order (one 32B contiguous load per lane per
//   wmma). Activations kept in f16; the 16x128 A tile is staged through LDS
//   (padded stride -> no bank conflicts) and shared by all 8 waves of a block.
//   Fused QKV reuses each A fragment for 3 wmma issues.
// - Edge phase: L2-resident gathers (node tables << 192MB L2) + f32 global
//   atomics for segment max / sum / scatter.
// ---------------------------------------------------------------------------

typedef __attribute__((ext_vector_type(16))) _Float16 v16h;
typedef __attribute__((ext_vector_type(8)))  _Float16 v8h;
typedef __attribute__((ext_vector_type(8)))  float    v8f;

static constexpr int N_NODES   = 50000;
static constexpr int E_EDGES   = 500000;
static constexpr int DIM       = 128;
static constexpr int HEADS     = 4;
static constexpr int HDIM      = 32;
static constexpr int ROW_TILES = N_NODES / 16;   // 3125, exact
static constexpr int LDS_STRIDE = DIM + 8;       // 136 halves: break 256B-stride bank conflicts

// ---------------- helpers ----------------

__device__ __forceinline__ unsigned ordf(float f) {
    unsigned u = __float_as_uint(f);
    return (u & 0x80000000u) ? ~u : (u | 0x80000000u);
}
__device__ __forceinline__ float deordf(unsigned o) {
    return (o & 0x80000000u) ? __uint_as_float(o ^ 0x80000000u)
                             : __uint_as_float(~o);
}
__device__ __forceinline__ float gelu_tanh(float x) {
    const float c0 = 0.7978845608028654f, c1 = 0.044715f;
    return 0.5f * x * (1.0f + tanhf(c0 * (x + c1 * x * x * x)));
}
__device__ __forceinline__ v16h joinfrag(v8h lo, v8h hi) {
    v16h a;
#pragma unroll
    for (int e = 0; e < 8; ++e) { a[e] = lo[e]; a[e + 8] = hi[e]; }
    return a;
}

// ---------------- weight pre-pack (one-time per layer) ----------------
// Dense 128x128 W -> f16 fragments: Wp[(kb*8+nt)*512 + lane*16 + e] =
//   W[(kb*32 + ((lane&16)>>1) + (e&7) + ((e&8)<<1))*128 + nt*16 + (lane&15)]
__global__ void pack_dense_w(const float* __restrict__ W, _Float16* __restrict__ Wp)
{
    const int idx  = blockIdx.x * blockDim.x + threadIdx.x;   // 0..16383
    const int tile = idx >> 9;
    const int lane = (idx >> 4) & 31;
    const int e    = idx & 15;
    const int kb = tile >> 3, nt = tile & 7;
    const int kk = kb * 32 + ((lane & 16) >> 1) + (e & 7) + ((e & 8) << 1);
    const int nn = nt * 16 + (lane & 15);
    Wp[idx] = (_Float16)W[kk * DIM + nn];
}

// Relation H x 32 x 32 -> f16 fragments: Rp[(h*2+c)*512 + lane*16 + e] =
//   R[h*1024 + (((lane&16)>>1) + (e&7) + ((e&8)<<1))*32 + c*16 + (lane&15)]
__global__ void pack_rel_w(const float* __restrict__ R, _Float16* __restrict__ Rp)
{
    const int idx  = blockIdx.x * blockDim.x + threadIdx.x;   // 0..4095
    const int frag = idx >> 9;
    const int lane = (idx >> 4) & 31;
    const int e    = idx & 15;
    const int h = frag >> 1, c = frag & 1;
    const int kk = ((lane & 16) >> 1) + (e & 7) + ((e & 8) << 1);
    const int nn = c * 16 + (lane & 15);
    Rp[idx] = (_Float16)R[h * HDIM * HDIM + kk * HDIM + nn];
}

// ---------------- elementwise converts ----------------
__global__ void conv_f16(const float* __restrict__ src, _Float16* __restrict__ dst, int n)
{
    const int i = (blockIdx.x * blockDim.x + threadIdx.x) * 4;
    if (i >= n) return;
    const float4 f = *(const float4*)(src + i);
    dst[i + 0] = (_Float16)f.x; dst[i + 1] = (_Float16)f.y;
    dst[i + 2] = (_Float16)f.z; dst[i + 3] = (_Float16)f.w;
}

__global__ void conv_gelu_f16(const float* __restrict__ src, _Float16* __restrict__ dst, int n)
{
    const int i = (blockIdx.x * blockDim.x + threadIdx.x) * 4;
    if (i >= n) return;
    const float4 f = *(const float4*)(src + i);
    dst[i + 0] = (_Float16)gelu_tanh(f.x); dst[i + 1] = (_Float16)gelu_tanh(f.y);
    dst[i + 2] = (_Float16)gelu_tanh(f.z); dst[i + 3] = (_Float16)gelu_tanh(f.w);
}

// ---------------- kernel 1: fused QKV GEMM (WMMA, LDS-staged A) ----------------
// grid (3125), block 256 (8 waves; wave == col tile). One LDS A-stage per
// block, each A fragment feeds 3 wmma (q,k,v).
__global__ void hgt_qkv_gemm(const _Float16* __restrict__ Xh,
                             const _Float16* __restrict__ Wqp, const float* __restrict__ bq,
                             const _Float16* __restrict__ Wkp, const float* __restrict__ bk,
                             const _Float16* __restrict__ Wvp, const float* __restrict__ bv,
                             float* __restrict__ Q,
                             _Float16* __restrict__ Kh, _Float16* __restrict__ Vh)
{
    __shared__ _Float16 As[16 * LDS_STRIDE];
    const int tid  = threadIdx.x;
    const int lane = tid & 31;
    const int nt   = tid >> 5;                    // 0..7 col tile
    const int m0   = blockIdx.x * 16;

    // cooperative stage: 16 rows x 128 halves, 8 halves per thread
    {
        const int row = tid >> 4, chunk = tid & 15;
        *(v8h*)&As[row * LDS_STRIDE + chunk * 8] =
            *(const v8h*)(Xh + (size_t)(m0 + row) * DIM + chunk * 8);
    }
    __syncthreads();

    const int aKoff = (lane & 16) >> 1;           // +8 halves for upper half-wave
    const int mloc  = (lane & 15) * LDS_STRIDE;

    v8f accq = {}, acck = {}, accv = {};
#pragma unroll
    for (int kb = 0; kb < 4; ++kb) {
        const v8h lo = *(const v8h*)&As[mloc + kb * 32 + aKoff];
        const v8h hi = *(const v8h*)&As[mloc + kb * 32 + 16 + aKoff];
        const v16h a = joinfrag(lo, hi);
        const int toff = (kb * 8 + nt) * 512 + lane * 16;
        const v16h bqf = *(const v16h*)(Wqp + toff);
        const v16h bkf = *(const v16h*)(Wkp + toff);
        const v16h bvf = *(const v16h*)(Wvp + toff);
        accq = __builtin_amdgcn_wmma_f32_16x16x32_f16(false, a, false, bqf, (short)0, accq, false, false);
        acck = __builtin_amdgcn_wmma_f32_16x16x32_f16(false, a, false, bkf, (short)0, acck, false, false);
        accv = __builtin_amdgcn_wmma_f32_16x16x32_f16(false, a, false, bvf, (short)0, accv, false, false);
    }

    const int nCol = nt * 16 + (lane & 15);
    const float bqn = bq[nCol], bkn = bk[nCol], bvn = bv[nCol];
#pragma unroll
    for (int r = 0; r < 8; ++r) {
        const int row = m0 + r + ((lane & 16) >> 1);
        Q [(size_t)row * DIM + nCol] = accq[r] + bqn;
        Kh[(size_t)row * DIM + nCol] = (_Float16)(acck[r] + bkn);
        Vh[(size_t)row * DIM + nCol] = (_Float16)(accv[r] + bvn);
    }
}

// ---------------- kernel 2: per-relation head transforms (WMMA) ----------------
// kr[n,h,:] = k[n,h,:] @ A[h];  vr[n,h,:] = v[n,h,:] @ M[h]
// grid (3125, 2): y = col half. block 128, wave == head. f16 A direct loads.
__global__ void hgt_rel_transform(const _Float16* __restrict__ Kh,
                                  const _Float16* __restrict__ Vh,
                                  const _Float16* __restrict__ Ap,
                                  const _Float16* __restrict__ Mp,
                                  float* __restrict__ KR, float* __restrict__ VR)
{
    const int lane = threadIdx.x & 31;
    const int h    = threadIdx.x >> 5;            // head 0..3
    const int m0   = blockIdx.x * 16;
    const int c    = blockIdx.y;                  // col half
    const int j0   = c * 16;

    const int aKoff = (lane & 16) >> 1;
    const size_t arow = (size_t)(m0 + (lane & 15)) * DIM + h * HDIM;

    const v16h ak = joinfrag(*(const v8h*)(Kh + arow + aKoff),
                             *(const v8h*)(Kh + arow + 16 + aKoff));
    const v16h av = joinfrag(*(const v8h*)(Vh + arow + aKoff),
                             *(const v8h*)(Vh + arow + 16 + aKoff));
    const int foff = (h * 2 + c) * 512 + lane * 16;
    const v16h ba = *(const v16h*)(Ap + foff);
    const v16h bm = *(const v16h*)(Mp + foff);

    v8f acck = {}, accv = {};
    acck = __builtin_amdgcn_wmma_f32_16x16x32_f16(false, ak, false, ba, (short)0, acck, false, false);
    accv = __builtin_amdgcn_wmma_f32_16x16x32_f16(false, av, false, bm, (short)0, accv, false, false);

    const int nCol = h * HDIM + j0 + (lane & 15);
#pragma unroll
    for (int r = 0; r < 8; ++r) {
        const int row = m0 + r + ((lane & 16) >> 1);
        KR[(size_t)row * DIM + nCol] = acck[r];
        VR[(size_t)row * DIM + nCol] = accv[r];
    }
}

// ---------------- kernel 3: edge logits + segment max ----------------
__global__ void hgt_edge_logits(const int* __restrict__ edges,
                                const float* __restrict__ Qm,
                                const float* __restrict__ KR,
                                const float* __restrict__ prel,
                                float* __restrict__ elog,
                                unsigned* __restrict__ mmax)
{
    const int e = blockIdx.x * blockDim.x + threadIdx.x;
    if (e >= E_EDGES) return;
    const int src = edges[e];
    const int dst = edges[E_EDGES + e];
    const float4* qr = (const float4*)(Qm + (size_t)dst * DIM);
    const float4* kr = (const float4*)(KR + (size_t)src * DIM);
    const float scale = 0.17677669529663687f;     // 1/sqrt(32)
#pragma unroll
    for (int h = 0; h < HEADS; ++h) {
        float acc = 0.0f;
#pragma unroll
        for (int j = 0; j < 8; ++j) {
            const float4 a = qr[h * 8 + j];
            const float4 b = kr[h * 8 + j];
            acc += a.x * b.x + a.y * b.y + a.z * b.z + a.w * b.w;
        }
        const float logit = acc * prel[h] * scale;
        elog[(size_t)e * HEADS + h] = logit;
        atomicMax(&mmax[(size_t)dst * HEADS + h], ordf(logit));
    }
}

// ---------------- kernel 4: exp + segment sum ----------------
__global__ void hgt_edge_exp(const int* __restrict__ edges,
                             const unsigned* __restrict__ mmax,
                             float* __restrict__ elog,
                             float* __restrict__ ssum)
{
    const int i = blockIdx.x * blockDim.x + threadIdx.x;
    if (i >= E_EDGES * HEADS) return;
    const int e = i >> 2;
    const int h = i & 3;
    const int dst = edges[E_EDGES + e];
    const float mx = deordf(mmax[(size_t)dst * HEADS + h]);
    const float ex = __expf(elog[i] - mx);
    elog[i] = ex;
    unsafeAtomicAdd(&ssum[(size_t)dst * HEADS + h], ex);
}

// ---------------- kernel 5: weighted message scatter ----------------
__global__ void hgt_edge_scatter(const int* __restrict__ edges,
                                 const float* __restrict__ elog,
                                 const float* __restrict__ ssum,
                                 const float* __restrict__ VR,
                                 float* __restrict__ att)
{
    const int i = blockIdx.x * blockDim.x + threadIdx.x;
    if (i >= E_EDGES * HEADS) return;
    const int e = i >> 2;
    const int h = i & 3;
    const int src = edges[e];
    const int dst = edges[E_EDGES + e];
    const float w = elog[i] / (ssum[(size_t)dst * HEADS + h] + 1e-16f);
    const float4* vr = (const float4*)(VR + (size_t)src * DIM + h * HDIM);
    float* ao = att + (size_t)dst * DIM + h * HDIM;
#pragma unroll
    for (int j = 0; j < 8; ++j) {
        const float4 v4 = vr[j];
        unsafeAtomicAdd(ao + j * 4 + 0, w * v4.x);
        unsafeAtomicAdd(ao + j * 4 + 1, w * v4.y);
        unsafeAtomicAdd(ao + j * 4 + 2, w * v4.z);
        unsafeAtomicAdd(ao + j * 4 + 3, w * v4.w);
    }
}

// ---------------- kernel 6: Wo GEMM + gated skip (WMMA, LDS-staged A) ----------
// A = f16(gelu(att)) precomputed. grid (3125), block 256 (wave == col tile).
__global__ void hgt_out_proj(const _Float16* __restrict__ Gh,
                             const _Float16* __restrict__ Wop,
                             const float* __restrict__ bo,
                             const float* __restrict__ skipP,
                             const float* __restrict__ Xin,
                             float* __restrict__ Hout)
{
    __shared__ _Float16 As[16 * LDS_STRIDE];
    const int tid  = threadIdx.x;
    const int lane = tid & 31;
    const int nt   = tid >> 5;
    const int m0   = blockIdx.x * 16;

    {
        const int row = tid >> 4, chunk = tid & 15;
        *(v8h*)&As[row * LDS_STRIDE + chunk * 8] =
            *(const v8h*)(Gh + (size_t)(m0 + row) * DIM + chunk * 8);
    }
    __syncthreads();

    const int aKoff = (lane & 16) >> 1;
    const int mloc  = (lane & 15) * LDS_STRIDE;

    v8f acc = {};
#pragma unroll
    for (int kb = 0; kb < 4; ++kb) {
        const v16h a = joinfrag(*(const v8h*)&As[mloc + kb * 32 + aKoff],
                                *(const v8h*)&As[mloc + kb * 32 + 16 + aKoff]);
        const v16h b = *(const v16h*)(Wop + (kb * 8 + nt) * 512 + lane * 16);
        acc = __builtin_amdgcn_wmma_f32_16x16x32_f16(false, a, false, b, (short)0, acc, false, false);
    }

    const float sg = 1.0f / (1.0f + __expf(-skipP[0]));
    const int nCol = nt * 16 + (lane & 15);
    const float bn = bo[nCol];
#pragma unroll
    for (int r = 0; r < 8; ++r) {
        const int row = m0 + r + ((lane & 16) >> 1);
        const float o = acc[r] + bn;
        Hout[(size_t)row * DIM + nCol] =
            sg * o + (1.0f - sg) * Xin[(size_t)row * DIM + nCol];
    }
}

// ---------------- launcher ----------------
// d_in: 0=x, 1=edges_like, 2=edges_retweet, 3=edges_follow, then param leaves
// in JAX pytree order (sorted dict keys), 18 per layer:
//  0 Wk  1 Wo  2 Wq  3 Wv  4 bk  5 bo  6 bq  7 bv
//  8 follow.A  9 follow.M 10 follow.p
// 11 like.A   12 like.M   13 like.p
// 14 retweet.A 15 retweet.M 16 retweet.p
// 17 skip
extern "C" void kernel_launch(void* const* d_in, const int* in_sizes, int n_in,
                              void* d_out, int out_size, void* d_ws, size_t ws_size,
                              hipStream_t stream)
{
    (void)in_sizes; (void)n_in; (void)out_size; (void)ws_size;

    const float* x = (const float*)d_in[0];
    const int* edges[3] = { (const int*)d_in[1],      // like
                            (const int*)d_in[2],      // retweet
                            (const int*)d_in[3] };    // follow
    const int relLeaf[3] = { 11, 14, 8 };             // A-leaf index per relation

    const size_t ND = (size_t)N_NODES * DIM;
    const size_t NH = (size_t)N_NODES * HEADS;
    const size_t EH = (size_t)E_EDGES * HEADS;

    float* q    = (float*)d_ws;
    float* kr   = q   + ND;
    float* vr   = kr  + ND;
    float* att  = vr  + ND;
    float* h1   = att + ND;
    unsigned* mbuf = (unsigned*)(h1 + ND);
    float* sbuf = (float*)(mbuf + NH);
    float* elog = sbuf + NH;
    _Float16* xh  = (_Float16*)(elog + EH);
    _Float16* kh  = xh + ND;
    _Float16* vh  = kh + ND;
    _Float16* gh  = vh + ND;
    _Float16* Wqp = gh + ND;        // 16384 halves each
    _Float16* Wkp = Wqp + 16384;
    _Float16* Wvp = Wkp + 16384;
    _Float16* Wop = Wvp + 16384;
    _Float16* Ap  = Wop + 16384;    // 4096 halves each
    _Float16* Mp  = Ap  + 4096;

    const int blkE = 256;
    const int nbE  = (E_EDGES + blkE - 1) / blkE;
    const int nbEH = (int)((EH + blkE - 1) / blkE);
    const int nbC  = (int)((ND / 4 + blkE - 1) / blkE);

    for (int L = 0; L < 2; ++L) {
        const int P = 4 + 18 * L;
        const float* Wk = (const float*)d_in[P + 0];
        const float* Wo = (const float*)d_in[P + 1];
        const float* Wq = (const float*)d_in[P + 2];
        const float* Wv = (const float*)d_in[P + 3];
        const float* bk = (const float*)d_in[P + 4];
        const float* bo = (const float*)d_in[P + 5];
        const float* bq = (const float*)d_in[P + 6];
        const float* bv = (const float*)d_in[P + 7];
        const float* skipP = (const float*)d_in[P + 17];

        const float* hin  = (L == 0) ? x  : h1;
        float*       hout = (L == 0) ? h1 : (float*)d_out;

        conv_f16<<<nbC, blkE, 0, stream>>>(hin, xh, (int)ND);
        pack_dense_w<<<64, 256, 0, stream>>>(Wq, Wqp);
        pack_dense_w<<<64, 256, 0, stream>>>(Wk, Wkp);
        pack_dense_w<<<64, 256, 0, stream>>>(Wv, Wvp);
        pack_dense_w<<<64, 256, 0, stream>>>(Wo, Wop);

        hgt_qkv_gemm<<<ROW_TILES, 256, 0, stream>>>(xh, Wqp, bq, Wkp, bk, Wvp, bv,
                                                    q, kh, vh);
        hipMemsetAsync(att, 0, ND * sizeof(float), stream);

        for (int r = 0; r < 3; ++r) {
            const float* Ar = (const float*)d_in[P + relLeaf[r] + 0];
            const float* Mr = (const float*)d_in[P + relLeaf[r] + 1];
            const float* pr = (const float*)d_in[P + relLeaf[r] + 2];

            pack_rel_w<<<16, 256, 0, stream>>>(Ar, Ap);
            pack_rel_w<<<16, 256, 0, stream>>>(Mr, Mp);
            hgt_rel_transform<<<dim3(ROW_TILES, 2), 128, 0, stream>>>(kh, vh, Ap, Mp, kr, vr);

            hipMemsetAsync(mbuf, 0, NH * sizeof(unsigned), stream);
            hipMemsetAsync(sbuf, 0, NH * sizeof(float), stream);
            hgt_edge_logits<<<nbE, blkE, 0, stream>>>(edges[r], q, kr, pr, elog, mbuf);
            hgt_edge_exp<<<nbEH, blkE, 0, stream>>>(edges[r], mbuf, elog, sbuf);
            hgt_edge_scatter<<<nbEH, blkE, 0, stream>>>(edges[r], elog, sbuf, vr, att);
        }

        conv_gelu_f16<<<nbC, blkE, 0, stream>>>(att, gh, (int)ND);
        hgt_out_proj<<<ROW_TILES, 256, 0, stream>>>(gh, Wop, bo, skipP, hin, hout);
    }
}
